// AttentionModel_19731079758444
// MI455X (gfx1250) — matrix operand order repaired and checked by reading, not compiled, not run on hardware
//
#include <hip/hip_runtime.h>
#include <math.h>

// ---------------------------------------------------------------------------
// Pointer-network (transformer encoder + LSTM pointer decoder) for MI455X.
// Heavy GEMMs use v_wmma_f32_16x16x32_f16 (wave32). Each GEMM wave now owns a
// 16x64 output strip (4 accumulators): 4 WMMAs per A-fragment load, doubling
// arithmetic intensity vs the 16x16-per-wave version.
// Attention is fused per (b,h) with the 160x160 score tile resident in LDS.
// Decoder scoring stays in f32 VALU (L2-bandwidth bound, argmax-sensitive).
// ---------------------------------------------------------------------------

typedef __attribute__((ext_vector_type(16))) _Float16 v16h;
typedef __attribute__((ext_vector_type(8)))  float    v8f;

#define Bsz   1024
#define Ntok  150
#define Edim  128
#define Hh    8
#define Dh    16
#define FFd   512
#define NLAY  3
#define ROWS  (Bsz*Ntok)      // 153600
#define NPAD  160

__device__ __forceinline__ float sigm_(float x){ return 1.f/(1.f+__expf(-x)); }

#define WMMA16(A_, B_, C_) \
  __builtin_amdgcn_wmma_f32_16x16x32_f16(false, (A_), false, (B_), (short)0, (C_), false, false)

// ------------------------------- converts ---------------------------------
__global__ void k_cvt16(const float* __restrict__ s, _Float16* __restrict__ d, int n){
  int i = blockIdx.x*blockDim.x + threadIdx.x;
  if (i < n) d[i] = (_Float16)s[i];
}

__global__ void k_wcat(const float* __restrict__ Wih, const float* __restrict__ Whh,
                       const float* __restrict__ bih, const float* __restrict__ bhh,
                       _Float16* __restrict__ Wc, float* __restrict__ bc){
  int i = blockIdx.x*blockDim.x + threadIdx.x;
  if (i < 512*256){
    int g = i >> 8, k = i & 255;
    float v = (k < 128) ? Wih[g*128 + k] : Whh[g*128 + (k-128)];
    Wc[i] = (_Float16)v;
  }
  if (i < 512) bc[i] = bih[i] + bhh[i];
}

// -------------------------- BN stats (deterministic) -----------------------
// input stats over the 4 raw channels: part[g*8 + {c, 4+c}]
__global__ void k_instats(const float* __restrict__ inp, float* __restrict__ part, int rows){
  int c = threadIdx.x & 3;
  int rl = threadIdx.x >> 2;                 // 0..31
  int step = (blockDim.x >> 2) * gridDim.x;
  float s = 0.f, s2 = 0.f;
  for (int r = blockIdx.x*(blockDim.x>>2) + rl; r < rows; r += step){
    float v = inp[(size_t)r*4 + c];
    s += v; s2 += v*v;
  }
  __shared__ float sh[128], sh2[128];
  sh[threadIdx.x] = s; sh2[threadIdx.x] = s2;
  __syncthreads();
  if (threadIdx.x < 4){
    float a = 0.f, a2 = 0.f;
    for (int j = threadIdx.x; j < 128; j += 4){ a += sh[j]; a2 += sh2[j]; }
    part[blockIdx.x*8 + threadIdx.x]     = a;
    part[blockIdx.x*8 + 4 + threadIdx.x] = a2;
  }
}

__global__ void k_finstats(const float* __restrict__ part, float* __restrict__ stats,
                           int twoC, int G){
  int j = threadIdx.x;
  if (j < twoC){
    float s = 0.f;
    for (int g = 0; g < G; g++) s += part[(size_t)g*twoC + j];
    stats[j] = s;
  }
}

// ---------------- embedding: BN(inp ch0,1)@opW + BN(ch2,3)@mtW -------------
__global__ void k_embed(const float* __restrict__ inp, const float* __restrict__ stIn,
                        const float* __restrict__ opg, const float* __restrict__ opb,
                        const float* __restrict__ opW, const float* __restrict__ opB,
                        const float* __restrict__ mtg, const float* __restrict__ mtb,
                        const float* __restrict__ mtW, const float* __restrict__ mtB,
                        float* __restrict__ r, float* __restrict__ part,
                        int rows, float invC){
  int e = threadIdx.x;  // 128 threads
  float m0 = stIn[0]*invC, m1 = stIn[1]*invC, m2 = stIn[2]*invC, m3 = stIn[3]*invC;
  float rs0 = rsqrtf(stIn[4]*invC - m0*m0 + 1e-5f)*opg[0];
  float rs1 = rsqrtf(stIn[5]*invC - m1*m1 + 1e-5f)*opg[1];
  float rs2 = rsqrtf(stIn[6]*invC - m2*m2 + 1e-5f)*mtg[0];
  float rs3 = rsqrtf(stIn[7]*invC - m3*m3 + 1e-5f)*mtg[1];
  float w00 = opW[e*2], w01 = opW[e*2+1], w10 = mtW[e*2], w11 = mtW[e*2+1];
  float bb = opB[e] + mtB[e];
  float s = 0.f, s2 = 0.f;
  for (int row = blockIdx.x; row < rows; row += gridDim.x){
    const float* x = inp + (size_t)row*4;
    float x0 = (x[0]-m0)*rs0 + opb[0];
    float x1 = (x[1]-m1)*rs1 + opb[1];
    float x2 = (x[2]-m2)*rs2 + mtb[0];
    float x3 = (x[3]-m3)*rs3 + mtb[1];
    float v = x0*w00 + x1*w01 + x2*w10 + x3*w11 + bb;
    r[(size_t)row*Edim + e] = v;
    s += v; s2 += v*v;
  }
  part[blockIdx.x*256 + e]       = s;
  part[blockIdx.x*256 + 128 + e] = s2;
}

// residual add + per-channel partial stats
__global__ void k_resid(const float* __restrict__ a, const float* __restrict__ y,
                        float* __restrict__ r, float* __restrict__ part, int rows){
  int e = threadIdx.x;  // 128
  float s = 0.f, s2 = 0.f;
  for (int row = blockIdx.x; row < rows; row += gridDim.x){
    size_t i = (size_t)row*Edim + e;
    float v = a[i] + y[i];
    r[i] = v; s += v; s2 += v*v;
  }
  part[blockIdx.x*256 + e]       = s;
  part[blockIdx.x*256 + 128 + e] = s2;
}

__global__ void k_bnapply(const float* __restrict__ r, const float* __restrict__ stats,
                          const float* __restrict__ g, const float* __restrict__ b,
                          float* __restrict__ h, _Float16* __restrict__ h16,
                          size_t n, float invC){
  size_t i = (size_t)blockIdx.x*blockDim.x + threadIdx.x;
  if (i >= n) return;
  int e = (int)(i & (Edim-1));
  float m  = stats[e]*invC;
  float va = stats[Edim+e]*invC - m*m;
  float sc = rsqrtf(va + 1e-5f)*g[e];
  float v  = (r[i]-m)*sc + b[e];
  h[i] = v;
  h16[i] = (_Float16)v;
}

// ----------------------- generic WMMA GEMM: C = A @ W^T + bias -------------
// A: MxK f16 row-major (stride lda). W: NxK f16 row-major.
// One 16x64 output strip per wave: 4 accumulators sharing one A fragment,
// so each 32-wide K chunk issues 4 v_wmma ops per 2 A-loads + 8 B-loads.
__global__ void k_gemm(const _Float16* __restrict__ A, int lda,
                       const _Float16* __restrict__ W,
                       const float* __restrict__ bias,
                       float* __restrict__ C32, _Float16* __restrict__ C16,
                       int M, int N, int K, int relu){
  int lane = threadIdx.x & 31;
  int wid  = threadIdx.x >> 5;
  int Ng = N >> 6;                           // 64-col groups (all N here %64==0)
  int tile = blockIdx.x*(blockDim.x >> 5) + wid;
  if (tile >= (M >> 4)*Ng) return;           // wave-uniform
  int mt = tile / Ng, ng = tile % Ng;
  int l15  = lane & 15, half = lane >> 4;
  const _Float16* pa  = A + (size_t)((mt<<4) + l15)*lda + half*16;
  const _Float16* pw0 = W + (size_t)((ng<<6) + l15)*K   + half*16;
  const size_t wstep = (size_t)16*K;         // next 16-col tile of W
  v8f acc0 = {}, acc1 = {}, acc2 = {}, acc3 = {};
  for (int kk = 0; kk < K; kk += 32){
    v16h a, b0, b1, b2, b3;
#pragma unroll
    for (int i = 0; i < 8; i++){ a[i] = pa[kk+i]; a[8+i] = pa[kk+8+i]; }
    const _Float16* pw = pw0 + kk;
#pragma unroll
    for (int i = 0; i < 16; i++){
      b0[i] = pw[i];
      b1[i] = pw[wstep + i];
      b2[i] = pw[2*wstep + i];
      b3[i] = pw[3*wstep + i];
    }
    acc0 = WMMA16(a, b0, acc0);
    acc1 = WMMA16(a, b1, acc1);
    acc2 = WMMA16(a, b2, acc2);
    acc3 = WMMA16(a, b3, acc3);
  }
  int rbase = (mt<<4) + (half<<3);
#pragma unroll
  for (int j = 0; j < 4; j++){
    v8f acc = (j==0) ? acc0 : (j==1) ? acc1 : (j==2) ? acc2 : acc3;
    int col = (ng<<6) + j*16 + l15;
    float bs = bias ? bias[col] : 0.f;
#pragma unroll
    for (int v = 0; v < 8; v++){
      float o = acc[v] + bs;
      if (relu) o = fmaxf(o, 0.f);
      size_t idx = (size_t)(rbase+v)*N + col;
      if (C32) C32[idx] = o;
      if (C16) C16[idx] = (_Float16)o;
    }
  }
}

// ------------- fused attention per (b,h): S=QK^T/4, softmax, P@V -----------
__global__ void k_attn(const float* __restrict__ qkv, _Float16* __restrict__ out16){
  __shared__ _Float16 S[NPAD*NPAD];           // 51.2 KB
  int b = blockIdx.x >> 3, h = blockIdx.x & 7;
  int lane = threadIdx.x & 31, wid = threadIdx.x >> 5;
  int l15 = lane & 15, half = lane >> 4;
  const float* qbase = qkv + (size_t)b*Ntok*(3*Edim) + h*Dh;

  // phase 1: 10x10 score tiles, K padded 16->32 with zeros
  for (int t = wid; t < 100; t += 8){
    int qt = t/10, kt = t%10;
    int qrow = qt*16 + l15, kcol = kt*16 + l15;
    v16h a, bb;
#pragma unroll
    for (int i = 0; i < 16; i++){ a[i] = (_Float16)0.f; bb[i] = (_Float16)0.f; }
    if (half == 0 && qrow < Ntok){
      const float* p = qbase + (size_t)qrow*(3*Edim);   // d = 0..15
#pragma unroll
      for (int i = 0; i < 16; i++) a[i] = (_Float16)p[i];
    }
    if (half == 0 && kcol < Ntok){
      const float* p = qbase + (size_t)kcol*(3*Edim) + Edim;     // K-block, d=0..15
#pragma unroll
      for (int i = 0; i < 16; i++) bb[i] = (_Float16)p[i];
    }
    v8f acc = {};
    acc = WMMA16(a, bb, acc);
    int col = kt*16 + l15, rbase = qt*16 + half*8;
#pragma unroll
    for (int v = 0; v < 8; v++)
      S[(rbase+v)*NPAD + col] = (_Float16)(acc[v]*0.25f);        // /sqrt(16)
  }
  __syncthreads();

  // phase 2: row softmax (one wave per row), zero padded cols
  for (int r = wid; r < Ntok; r += 8){
    float m = -1e30f;
    for (int n = lane; n < Ntok; n += 32) m = fmaxf(m, (float)S[r*NPAD+n]);
    for (int o = 16; o; o >>= 1) m = fmaxf(m, __shfl_down(m, o));
    m = __shfl(m, 0);
    float s = 0.f;
    for (int n = lane; n < Ntok; n += 32) s += __expf((float)S[r*NPAD+n] - m);
    for (int o = 16; o; o >>= 1) s += __shfl_down(s, o);
    s = __shfl(s, 0);
    float inv = 1.f/s;
    for (int n = lane; n < Ntok; n += 32)
      S[r*NPAD+n] = (_Float16)(__expf((float)S[r*NPAD+n] - m)*inv);
    if (lane < NPAD-Ntok) S[r*NPAD + Ntok + lane] = (_Float16)0.f;
  }
  __syncthreads();

  // phase 3: O = P @ V, K = 160 in five 32-chunks
  const float* vbase = qkv + (size_t)b*Ntok*(3*Edim) + 2*Edim + h*Dh;
  for (int mt = wid; mt < 10; mt += 8){
    int prow = mt*16 + l15;
    v8f acc = {};
    for (int kc = 0; kc < 5; kc++){
      v16h a, bb;
#pragma unroll
      for (int i = 0; i < 8; i++){
        a[i]   = S[prow*NPAD + kc*32 + half*8 + i];
        a[8+i] = S[prow*NPAD + kc*32 + 16 + half*8 + i];
      }
#pragma unroll
      for (int i = 0; i < 16; i++){
        int krow = kc*32 + ((i < 8) ? (half*8 + i) : (16 + half*8 + (i - 8)));
        bb[i] = (krow < Ntok) ? (_Float16)vbase[(size_t)krow*(3*Edim) + l15]
                              : (_Float16)0.f;
      }
      acc = WMMA16(a, bb, acc);
    }
    int rbase = mt*16 + half*8;
#pragma unroll
    for (int v = 0; v < 8; v++){
      int n = rbase + v;
      if (n < Ntok)
        out16[((size_t)b*Ntok + n)*Edim + h*Dh + l15] = (_Float16)acc[v];
    }
  }
}

// ----------------------------- decoder pieces ------------------------------
__global__ void k_encmean(const float* __restrict__ hF, float* __restrict__ cc){
  int b = blockIdx.x, e = threadIdx.x;
  float s = 0.f;
  for (int n = 0; n < Ntok; n++) s += hF[((size_t)b*Ntok + n)*Edim + e];
  cc[b*Edim + e] = s * (1.f/Ntok);
}

__global__ void k_nainit(const float* __restrict__ inp, unsigned char* __restrict__ na,
                         int total){
  int i = blockIdx.x*blockDim.x + threadIdx.x;
  if (i < total) na[i] = (inp[(size_t)i*4 + 1] != 0.f) ? 1 : 0;
}

__global__ void k_lstm(const float* __restrict__ gates, float* __restrict__ cc,
                       _Float16* __restrict__ xcat){
  int i = blockIdx.x*blockDim.x + threadIdx.x;
  if (i >= Bsz*Edim) return;
  int b = i >> 7, e = i & 127;
  const float* g = gates + (size_t)b*512;
  float gi = g[e], gf = g[128+e], gg = g[256+e], go = g[384+e];
  float c  = sigm_(gf)*cc[i] + sigm_(gi)*tanhf(gg);
  float hh = sigm_(go)*tanhf(c);
  cc[i] = c;
  xcat[(size_t)b*256 + 128 + e] = (_Float16)hh;   // hh slot for next GEMMs
}

// one pointer-attention step for batch row b (block per b)
__global__ void k_ptr(const float* __restrict__ encw1, const float* __restrict__ u,
                      const float* __restrict__ pv, const float* __restrict__ pbv,
                      const float* __restrict__ enc, _Float16* __restrict__ xcat,
                      unsigned char* __restrict__ sch, unsigned char* __restrict__ na,
                      float* __restrict__ lls, int* __restrict__ seq, int t){
  int b = blockIdx.x;
  int lane = threadIdx.x & 31, wid = threadIdx.x >> 5;
  __shared__ float a_sh[NPAD];
  __shared__ float mSh, sSh; __shared__ int predSh;

  for (int n = wid; n < Ntok; n += 8){
    float acc = 0.f;
    const float* pe = encw1 + ((size_t)b*Ntok + n)*Edim;
    const float* pu = u + (size_t)b*Edim;
    for (int e = lane; e < Edim; e += 32)
      acc += tanhf(pe[e] + pu[e]) * pv[e];
    for (int o = 16; o; o >>= 1) acc += __shfl_down(acc, o);
    if (lane == 0){
      bool masked = sch[b*Ntok + n] || na[b*Ntok + n];
      a_sh[n] = masked ? -10000.f : (acc + pbv[0]);
    }
  }
  __syncthreads();
  if (wid == 0){
    float m = -1e30f; int pred = 0;
    for (int n = lane; n < Ntok; n += 32){
      float v = a_sh[n];
      if (v > m){ m = v; pred = n; }
    }
    for (int o = 16; o; o >>= 1){
      float om = __shfl_down(m, o); int op = __shfl_down(pred, o);
      if (om > m || (om == m && op < pred)){ m = om; pred = op; }
    }
    m = __shfl(m, 0); pred = __shfl(pred, 0);
    float s = 0.f;
    for (int n = lane; n < Ntok; n += 32) s += __expf(a_sh[n] - m);
    for (int o = 16; o; o >>= 1) s += __shfl_down(s, o);
    if (lane == 0){ mSh = m; sSh = s; predSh = pred; }
  }
  __syncthreads();
  int pred = predSh;
  for (int e = threadIdx.x; e < Edim; e += blockDim.x)       // dd = enc[b,pred]
    xcat[(size_t)b*256 + e] = (_Float16)enc[((size_t)b*Ntok + pred)*Edim + e];
  if (threadIdx.x == 0){
    seq[b*Ntok + t] = pred;
    sch[b*Ntok + pred] = 1;
    int nn = pred + 1; if (nn > Ntok-1) nn = Ntok-1;
    na[b*Ntok + nn] = 0;
    lls[b] += a_sh[pred] - mSh - __logf(sSh);
  }
}

__global__ void k_out(const float* __restrict__ inp, const int* __restrict__ seq,
                      const float* __restrict__ lls, float* __restrict__ out){
  int i = blockIdx.x*blockDim.x + threadIdx.x;
  if (i < ROWS){
    int b = i / Ntok;
    int s = seq[i];
    const float* src = inp + ((size_t)b*Ntok + s)*4;
    float* dst = out + (size_t)i*4;
    dst[0] = src[0]; dst[1] = src[1]; dst[2] = src[2]; dst[3] = src[3];
  }
  if (i < Bsz) out[(size_t)ROWS*4 + i] = lls[i];
}

// ---------------------------------------------------------------------------
extern "C" void kernel_launch(void* const* d_in, const int* in_sizes, int n_in,
                              void* d_out, int out_size, void* d_ws, size_t ws_size,
                              hipStream_t stream) {
  const float* inp   = (const float*)d_in[0];
  const float* opg   = (const float*)d_in[1];
  const float* opbb  = (const float*)d_in[2];
  const float* opW   = (const float*)d_in[3];
  const float* opB   = (const float*)d_in[4];
  const float* mtg   = (const float*)d_in[5];
  const float* mtbb  = (const float*)d_in[6];
  const float* mtW   = (const float*)d_in[7];
  const float* mtB   = (const float*)d_in[8];
  const float* enc0g = (const float*)d_in[9];
  const float* enc0b = (const float*)d_in[10];
  const float* inw   = (const float*)d_in[11];
  const float* inb   = (const float*)d_in[12];
  const float* outw  = (const float*)d_in[13];
  const float* outb  = (const float*)d_in[14];
  const float* bn1g  = (const float*)d_in[15];
  const float* bn1b  = (const float*)d_in[16];
  const float* ffw1  = (const float*)d_in[17];
  const float* ffb1  = (const float*)d_in[18];
  const float* ffw2  = (const float*)d_in[19];
  const float* ffb2  = (const float*)d_in[20];
  const float* bn2g  = (const float*)d_in[21];
  const float* bn2b  = (const float*)d_in[22];
  const float* Wih   = (const float*)d_in[23];
  const float* Whh   = (const float*)d_in[24];
  const float* bih   = (const float*)d_in[25];
  const float* bhh   = (const float*)d_in[26];
  const float* pw1   = (const float*)d_in[27];
  const float* pb1   = (const float*)d_in[28];
  const float* pw2   = (const float*)d_in[29];
  const float* pb2   = (const float*)d_in[30];
  const float* pv    = (const float*)d_in[31];
  const float* pbv   = (const float*)d_in[32];
  float* out = (float*)d_out;

  // ----- workspace carve (aligned 256B) -----
  char* base = (char*)d_ws; size_t off = 0;
  auto carve = [&](size_t bytes)->void*{
    void* p = base + off; off += (bytes + 255) & ~(size_t)255; return p;
  };
  float*    hF     = (float*)   carve((size_t)ROWS*Edim*4);
  float*    rF     = (float*)   carve((size_t)ROWS*Edim*4);
  float*    yF     = (float*)   carve((size_t)ROWS*Edim*4);
  float*    qkvF   = (float*)   carve((size_t)ROWS*3*Edim*4);  // aliased: encw1 later
  _Float16* h16    = (_Float16*)carve((size_t)ROWS*Edim*2);
  _Float16* attn16 = (_Float16*)carve((size_t)ROWS*Edim*2);
  _Float16* ff116  = (_Float16*)carve((size_t)ROWS*FFd*2);
  _Float16* inw16  = (_Float16*)carve((size_t)NLAY*3*Edim*Edim*2);
  _Float16* outw16 = (_Float16*)carve((size_t)NLAY*Edim*Edim*2);
  _Float16* f1w16  = (_Float16*)carve((size_t)NLAY*FFd*Edim*2);
  _Float16* f2w16  = (_Float16*)carve((size_t)NLAY*Edim*FFd*2);
  _Float16* pw116  = (_Float16*)carve((size_t)Edim*Edim*2);
  _Float16* pw216  = (_Float16*)carve((size_t)Edim*Edim*2);
  _Float16* wcat16 = (_Float16*)carve((size_t)512*256*2);
  float*    bcat   = (float*)   carve(512*4);
  float*    part   = (float*)   carve((size_t)256*256*4);
  float*    stats  = (float*)   carve(256*4);
  float*    statsI = (float*)   carve(8*4);
  float*    ccF    = (float*)   carve((size_t)Bsz*Edim*4);
  float*    uF     = (float*)   carve((size_t)Bsz*Edim*4);
  float*    gatesF = (float*)   carve((size_t)Bsz*512*4);
  _Float16* xcat   = (_Float16*)carve((size_t)Bsz*256*2);
  float*    llsF   = (float*)   carve(Bsz*4);
  int*      seqI   = (int*)     carve((size_t)ROWS*4);
  unsigned char* schB = (unsigned char*)carve(ROWS);
  unsigned char* naB  = (unsigned char*)carve(ROWS);
  float* encw1F = qkvF;   // qkv buffer is dead when the decoder starts

  const float invC = 1.f/(float)ROWS;
  auto gemmL = [&](const _Float16* A, int lda, const _Float16* W, const float* bias,
                   float* C32, _Float16* C16, int M, int Nn, int K, int relu){
    int tiles = (M/16)*(Nn/64);              // one 16x64 strip per wave
    k_gemm<<<(tiles+7)/8, 256, 0, stream>>>(A, lda, W, bias, C32, C16, M, Nn, K, relu);
  };

  // ----- weight conversion (f32 -> f16) -----
  { int n = NLAY*3*Edim*Edim; k_cvt16<<<(n+255)/256,256,0,stream>>>(inw,  inw16,  n); }
  { int n = NLAY*Edim*Edim;   k_cvt16<<<(n+255)/256,256,0,stream>>>(outw, outw16, n); }
  { int n = NLAY*FFd*Edim;    k_cvt16<<<(n+255)/256,256,0,stream>>>(ffw1, f1w16,  n); }
  { int n = NLAY*Edim*FFd;    k_cvt16<<<(n+255)/256,256,0,stream>>>(ffw2, f2w16,  n); }
  { int n = Edim*Edim;        k_cvt16<<<(n+255)/256,256,0,stream>>>(pw1,  pw116,  n); }
  { int n = Edim*Edim;        k_cvt16<<<(n+255)/256,256,0,stream>>>(pw2,  pw216,  n); }
  k_wcat<<<(512*256+255)/256, 256, 0, stream>>>(Wih, Whh, bih, bhh, wcat16, bcat);

  // ----- embedding + enc0 BN -----
  k_instats<<<256,128,0,stream>>>(inp, part, ROWS);
  k_finstats<<<1,256,0,stream>>>(part, statsI, 8, 256);
  k_embed<<<256,128,0,stream>>>(inp, statsI, opg, opbb, opW, opB, mtg, mtbb, mtW, mtB,
                                rF, part, ROWS, invC);
  k_finstats<<<1,256,0,stream>>>(part, stats, 256, 256);
  {
    size_t n = (size_t)ROWS*Edim;
    k_bnapply<<<(unsigned)((n+255)/256),256,0,stream>>>(rF, stats, enc0g, enc0b,
                                                        hF, h16, n, invC);
  }

  // ----- 3 transformer encoder layers -----
  for (int l = 0; l < NLAY; l++){
    gemmL(h16, Edim, inw16 + (size_t)l*3*Edim*Edim, inb + l*3*Edim,
          qkvF, nullptr, ROWS, 3*Edim, Edim, 0);
    k_attn<<<Bsz*Hh, 256, 0, stream>>>(qkvF, attn16);
    gemmL(attn16, Edim, outw16 + (size_t)l*Edim*Edim, outb + l*Edim,
          yF, nullptr, ROWS, Edim, Edim, 0);
    k_resid<<<256,128,0,stream>>>(hF, yF, rF, part, ROWS);
    k_finstats<<<1,256,0,stream>>>(part, stats, 256, 256);
    { size_t n=(size_t)ROWS*Edim;
      k_bnapply<<<(unsigned)((n+255)/256),256,0,stream>>>(rF, stats, bn1g+l*Edim,
                                                          bn1b+l*Edim, hF, h16, n, invC); }
    gemmL(h16, Edim, f1w16 + (size_t)l*FFd*Edim, ffb1 + l*FFd,
          nullptr, ff116, ROWS, FFd, Edim, 1);
    gemmL(ff116, FFd, f2w16 + (size_t)l*Edim*FFd, ffb2 + l*Edim,
          yF, nullptr, ROWS, Edim, FFd, 0);
    k_resid<<<256,128,0,stream>>>(hF, yF, rF, part, ROWS);
    k_finstats<<<1,256,0,stream>>>(part, stats, 256, 256);
    { size_t n=(size_t)ROWS*Edim;
      k_bnapply<<<(unsigned)((n+255)/256),256,0,stream>>>(rF, stats, bn2g+l*Edim,
                                                          bn2b+l*Edim, hF, h16, n, invC); }
  }

  // ----- decoder prep -----
  gemmL(h16, Edim, pw116, pb1, encw1F, nullptr, ROWS, Edim, Edim, 0);  // enc @ pw1^T + pb1
  k_encmean<<<Bsz,128,0,stream>>>(hF, ccF);                            // c0
  hipMemsetAsync(xcat, 0, (size_t)Bsz*256*2, stream);                  // h0 = d0 = 0
  hipMemsetAsync(llsF, 0, Bsz*4, stream);
  hipMemsetAsync(schB, 0, ROWS, stream);
  k_nainit<<<(ROWS+255)/256,256,0,stream>>>(inp, naB, ROWS);

  // ----- 150 sequential decode steps -----
  for (int t = 0; t < Ntok; t++){
    gemmL(xcat,     256, wcat16, bcat, gatesF, nullptr, Bsz, 512, 256, 0);  // LSTM gates
    k_lstm<<<(Bsz*Edim+255)/256,256,0,stream>>>(gatesF, ccF, xcat);
    gemmL(xcat+128, 256, pw216,  pb2,  uF,     nullptr, Bsz, Edim, Edim, 0); // u = hh@pw2^T
    k_ptr<<<Bsz,256,0,stream>>>(encw1F, uF, pv, pbv, hF, xcat, schB, naB,
                                llsF, seqI, t);
  }

  // ----- gather output sequence + lls -----
  k_out<<<(ROWS+255)/256,256,0,stream>>>(inp, seqI, llsF, out);
  (void)in_sizes; (void)n_in; (void)out_size; (void)ws_size;
}
